// Graph3DBias_51531017617624
// MI455X (gfx1250) — matrix operand order repaired and checked by reading, not compiled, not run on hardware
//
#include <hip/hip_runtime.h>
#include <hip/hip_bf16.h>
#include <math.h>

typedef _Float16 h16;
typedef __attribute__((ext_vector_type(16))) _Float16 v16h;
typedef __attribute__((ext_vector_type(8)))  _Float16 v8h;
typedef __attribute__((ext_vector_type(8)))  float    v8f;

#define B_ 2
#define N_ 512
#define FX_ 9
#define K_ 128
#define H_ 32
#define D_ 768

// output layout (floats): gab [B,H,N,N] | merge [B,N,D] | delta_pos [B,N,N,3]
#define MERGE_OFF (B_*H_*N_*N_)           // 16777216
#define DP_OFF    (MERGE_OFF + B_*N_*D_)  // 17563648

// workspace layout (bytes)
#define WS_SUM   0           // B*N*K f32      = 524288
#define WS_MASK  524288      // B*N int32      = 4096
#define WS_W1B   528384      // 16384 h16      = 32768
#define WS_W2B   561152      // 4096 h16       = 8192
#define WS_WPB   569344      // 98304 h16      = 196608

// Branch-free exact-gelu: 0.5*x*(1+erf(x/sqrt(2))) with Abramowitz–Stegun
// 7.1.26 erf (|err| < 1.5e-7).  Lowers to v_rcp_f32 + v_exp_f32 + FMAs,
// no exec-mask divergence (unlike libm erff).
__device__ __forceinline__ float gelu_exact(float x)
{
    float ax  = fabsf(x) * 0.70710678f;                    // |x|/sqrt(2)
    float t   = __builtin_amdgcn_rcpf(1.0f + 0.3275911f * ax);
    float p   = t*(0.254829592f + t*(-0.284496736f + t*(1.421413741f
               + t*(-1.453152027f + t*1.061405429f))));
    float erfv = 1.0f - p * __expf(-ax*ax);
    erfv = copysignf(erfv, x);
    return 0.5f * x * (1.0f + erfv);
}

// ---------------------------------------------------------------------------
// Prep: zero column-sum buffer, padding mask, pre-swizzle W1/W2/Wp into
// WMMA B-fragment order: frag[(s*NT+nt)*32+lane][q], k = s*32 + (lane>>4)*8 + (q&7) + (q>>3)*16,
// col = nt*16 + (lane&15).
// ---------------------------------------------------------------------------
__global__ __launch_bounds__(256)
void prep_kernel(const int* __restrict__ x, const float* __restrict__ W1,
                 const float* __restrict__ W2, const float* __restrict__ Wp,
                 float* __restrict__ sumEdge, int* __restrict__ maskArr,
                 h16* __restrict__ w1b, h16* __restrict__ w2b, h16* __restrict__ wpb)
{
    int tid = blockIdx.x * blockDim.x + threadIdx.x;
    if (tid < B_*N_*K_) { sumEdge[tid] = 0.0f; return; }
    tid -= B_*N_*K_;
    if (tid < B_*N_) {
        const int* xr = x + tid * FX_;
        int nz = 0;
        #pragma unroll
        for (int f = 0; f < FX_; ++f) nz |= xr[f];
        maskArr[tid] = (nz == 0) ? 1 : 0;
        return;
    }
    tid -= B_*N_;
    if (tid < 4*8*32*16) {                         // W1 [128,128]
        int q = tid & 15, l = (tid >> 4) & 31, nt = (tid >> 9) & 7, s = tid >> 12;
        int k   = s*32 + ((l >> 4) << 3) + (q & 7) + ((q >> 3) << 4);
        int col = nt*16 + (l & 15);
        w1b[tid] = (h16)W1[k*K_ + col];
        return;
    }
    tid -= 4*8*32*16;
    if (tid < 4*2*32*16) {                         // W2 [128,32]
        int q = tid & 15, l = (tid >> 4) & 31, nt = (tid >> 9) & 1, s = tid >> 10;
        int k   = s*32 + ((l >> 4) << 3) + (q & 7) + ((q >> 3) << 4);
        int col = nt*16 + (l & 15);
        w2b[tid] = (h16)W2[k*H_ + col];
        return;
    }
    tid -= 4*2*32*16;
    if (tid < 4*48*32*16) {                        // Wp [128,768]
        int q = tid & 15; int rest = tid >> 4; int l = rest & 31; rest >>= 5;
        int nt = rest % 48; int s = rest / 48;
        int k   = s*32 + ((l >> 4) << 3) + (q & 7) + ((q >> 3) << 4);
        int col = nt*16 + (l & 15);
        wpb[tid] = (h16)Wp[k*D_ + col];
    }
}

// ---------------------------------------------------------------------------
// Main fused edge kernel: one block = (b, i, 128 consecutive j).
// ---------------------------------------------------------------------------
__global__ __launch_bounds__(256)
void edge_kernel(const int* __restrict__ nte, const float* __restrict__ pos,
                 const float* __restrict__ means, const float* __restrict__ stds,
                 const float* __restrict__ mul_table, const float* __restrict__ bias_table,
                 const float* __restrict__ b1, const float* __restrict__ b2,
                 const h16* __restrict__ w1b, const h16* __restrict__ w2b,
                 const int* __restrict__ maskArr, float* __restrict__ sumEdge,
                 float* __restrict__ out)
{
    __shared__ __align__(16) h16 efh[128*136];    // edge features, f16
    __shared__ __align__(16) h16 hmidh[128*72];   // gelu(GEMM1) half-tile
    __shared__ float gArr[128];
    __shared__ float mjArr[128];
    __shared__ float kmean[128], kinv[128], kcoef[128];
    __shared__ float pArr[256];                   // masked-sum partials

    const int tile = blockIdx.x;
    const int j0 = (tile & 3) * 128;
    const int i  = (tile >> 2) & (N_ - 1);
    const int b  = tile >> 11;
    const int t  = threadIdx.x;

    // warm the L2/L0 lines of the pre-swizzled weights while we do scalar work
    __builtin_prefetch(w1b + t*64, 0, 0);         // 16384 halves
    __builtin_prefetch(w2b + t*16, 0, 0);         // 4096 halves

    // ---- phase 1: per-edge scalars + delta_pos output; per-k constants ----
    if (t < 128) {
        int j = j0 + t;
        const float* pb = pos + (size_t)b * N_ * 3;
        float pix = pb[i*3+0], piy = pb[i*3+1], piz = pb[i*3+2];
        float dx = pb[j*3+0] - pix, dy = pb[j*3+1] - piy, dz = pb[j*3+2] - piz;
        float dist = sqrtf(dx*dx + dy*dy + dz*dz);
        float inv  = 1.0f / (dist + 1e-5f);
        int dpi = DP_OFF + ((b*N_ + i)*N_ + j)*3;
        out[dpi+0] = dx*inv; out[dpi+1] = dy*inv; out[dpi+2] = dz*inv;
        int eb = ((b*N_ + i)*N_ + j)*2;
        int e0 = nte[eb], e1 = nte[eb+1];
        float mul = mul_table[e0] + mul_table[e1];
        float bia = bias_table[e0] + bias_table[e1];
        gArr[t]  = mul*dist + bia;
        mjArr[t] = maskArr[b*N_ + j] ? 0.0f : 1.0f;
    } else {
        int k = t - 128;
        float sd = fabsf(stds[k]) + 0.01f;
        kmean[k] = means[k];
        kinv[k]  = 1.0f / sd;
        kcoef[k] = 0.3989424f / sd;   // 1/(sqrt(2*3.14159)*std)
    }
    __syncthreads();

    // ---- phase 2: Gaussian RBF -> f16 LDS, fused j-masked column sums ----
    {
        const int k = t & 127;
        const int mb = t >> 7;                    // even/odd m split
        const float mean = kmean[k], kiv = kinv[k], kco = kcoef[k];
        float partial = 0.0f;
        #pragma unroll 8
        for (int n = 0; n < 64; ++n) {
            int m = mb + 2*n;
            float tt = (gArr[m] - mean) * kiv;
            float e = __expf(-0.5f*tt*tt) * kco;
            efh[m*136 + k] = (h16)e;
            partial += mjArr[m] * e;
        }
        pArr[t] = partial;
    }
    __syncthreads();

    if (t < 128)
        unsafeAtomicAdd(&sumEdge[(b*N_ + i)*K_ + t], pArr[t] + pArr[t + 128]);

    // ---- phase 3: WMMA GEMM1 (gelu) -> GEMM2 ----
    const int wv = t >> 5, lane = t & 31;
    const int mrow = wv*16 + (lane & 15);
    const int kseg = lane >> 4;

    v16h afrag[4];
    #pragma unroll
    for (int s = 0; s < 4; ++s) {
        const h16* base = efh + mrow*136 + s*32 + kseg*8;
        v8h lo = *(const v8h*)(base);
        v8h hi = *(const v8h*)(base + 16);
        #pragma unroll
        for (int q = 0; q < 8; ++q) { afrag[s][q] = lo[q]; afrag[s][q+8] = hi[q]; }
    }

    const v16h* w1f = (const v16h*)w1b;
    const v16h* w2f = (const v16h*)w2b;

    v8f acc2[2];
    acc2[0] = (v8f){0,0,0,0,0,0,0,0};
    acc2[1] = acc2[0];

    #pragma unroll
    for (int ch = 0; ch < 2; ++ch) {
        #pragma unroll
        for (int ct = 0; ct < 4; ++ct) {
            int ctg = ch*4 + ct;
            v8f c1 = {};
            #pragma unroll
            for (int s = 0; s < 4; ++s) {
                v16h bfrag = w1f[(s*8 + ctg)*32 + lane];
                c1 = __builtin_amdgcn_wmma_f32_16x16x32_f16(false, afrag[s], false, bfrag,
                                                            (short)0, c1, false, false);
            }
            int ncol = ct*16 + (lane & 15);
            float bb = b1[ch*64 + ncol];
            #pragma unroll
            for (int r = 0; r < 8; ++r) {
                int ml = wv*16 + ((lane < 16) ? r : 8 + r);
                hmidh[ml*72 + ncol] = (h16)gelu_exact(c1[r] + bb);
            }
        }
        #pragma unroll
        for (int s2 = 0; s2 < 2; ++s2) {
            const h16* base = hmidh + mrow*72 + s2*32 + kseg*8;
            v8h lo = *(const v8h*)(base);
            v8h hi = *(const v8h*)(base + 16);
            v16h a2;
            #pragma unroll
            for (int q = 0; q < 8; ++q) { a2[q] = lo[q]; a2[q+8] = hi[q]; }
            int sg = ch*2 + s2;
            #pragma unroll
            for (int c = 0; c < 2; ++c) {
                v16h bfrag = w2f[(sg*2 + c)*32 + lane];
                acc2[c] = __builtin_amdgcn_wmma_f32_16x16x32_f16(false, a2, false, bfrag,
                                                                 (short)0, acc2[c], false, false);
            }
        }
    }

    // ---- epilogue: gab[b,h,i,j], -inf j-mask, packed b128 stores ----
    const int mb8 = wv*16 + ((lane < 16) ? 0 : 8);   // 8 consecutive j per lane
    #pragma unroll
    for (int c = 0; c < 2; ++c) {
        int h = c*16 + (lane & 15);
        float bbv = b2[h];
        float v[8];
        #pragma unroll
        for (int r = 0; r < 8; ++r) {
            float val = acc2[c][r] + bbv;
            if (mjArr[mb8 + r] == 0.0f) val = -INFINITY;
            v[r] = val;
        }
        float* dst = out + ((size_t)((b*H_ + h)*N_ + i))*N_ + j0 + mb8;
        *(float4*)(dst)     = make_float4(v[0], v[1], v[2], v[3]);
        *(float4*)(dst + 4) = make_float4(v[4], v[5], v[6], v[7]);
    }
}

// ---------------------------------------------------------------------------
// merge_edge_features = sum_edge @ Wp + bp, i-masked.  [1024,128]@[128,768]
// ---------------------------------------------------------------------------
__global__ __launch_bounds__(256)
void merge_kernel(const float* __restrict__ sumEdge, const h16* __restrict__ wpb,
                  const float* __restrict__ bp, const int* __restrict__ maskArr,
                  float* __restrict__ out)
{
    const int rb = blockIdx.x;               // 16-row tile of [B*N]
    const int t  = threadIdx.x;
    const int wv = t >> 5, lane = t & 31;
    const int row  = rb*16 + (lane & 15);
    const int kseg = lane >> 4;

    v16h afrag[4];
    #pragma unroll
    for (int s = 0; s < 4; ++s) {
        const float* base = sumEdge + (size_t)row*K_ + s*32 + kseg*8;
        #pragma unroll
        for (int q = 0; q < 8; ++q) {
            afrag[s][q]   = (h16)base[q];
            afrag[s][q+8] = (h16)base[16 + q];
        }
    }

    const v16h* wpf = (const v16h*)wpb;
    #pragma unroll
    for (int u = 0; u < 6; ++u) {
        int nt = wv*6 + u;                    // 48 col tiles / 8 waves
        v8f acc = {};
        #pragma unroll
        for (int s = 0; s < 4; ++s) {
            v16h bf = wpf[(s*48 + nt)*32 + lane];
            acc = __builtin_amdgcn_wmma_f32_16x16x32_f16(false, afrag[s], false, bf,
                                                         (short)0, acc, false, false);
        }
        int h = nt*16 + (lane & 15);
        float bpv = bp[h];
        #pragma unroll
        for (int r = 0; r < 8; ++r) {
            int m = rb*16 + ((lane < 16) ? r : 8 + r);
            float val = maskArr[m] ? 0.0f : (acc[r] + bpv);
            out[MERGE_OFF + (size_t)m*D_ + h] = val;
        }
    }
}

// ---------------------------------------------------------------------------
extern "C" void kernel_launch(void* const* d_in, const int* in_sizes, int n_in,
                              void* d_out, int out_size, void* d_ws, size_t ws_size,
                              hipStream_t stream)
{
    const int*   x     = (const int*)  d_in[0];
    const int*   nte   = (const int*)  d_in[1];
    const float* pos   = (const float*)d_in[2];
    const float* means = (const float*)d_in[3];
    const float* stds  = (const float*)d_in[4];
    const float* mult  = (const float*)d_in[5];
    const float* biat  = (const float*)d_in[6];
    const float* W1    = (const float*)d_in[7];
    const float* b1    = (const float*)d_in[8];
    const float* W2    = (const float*)d_in[9];
    const float* b2    = (const float*)d_in[10];
    const float* Wp    = (const float*)d_in[11];
    const float* bp    = (const float*)d_in[12];
    float* out = (float*)d_out;

    char* ws = (char*)d_ws;
    float* sumEdge = (float*)(ws + WS_SUM);
    int*   maskArr = (int*)  (ws + WS_MASK);
    h16*   w1b     = (h16*)  (ws + WS_W1B);
    h16*   w2b     = (h16*)  (ws + WS_W2B);
    h16*   wpb     = (h16*)  (ws + WS_WPB);

    // total prep items: 131072 + 1024 + 16384 + 4096 + 98304 = 250880
    prep_kernel<<<(250880 + 255)/256, 256, 0, stream>>>(x, W1, W2, Wp,
                                                        sumEdge, maskArr, w1b, w2b, wpb);
    edge_kernel<<<B_*N_*(N_/128), 256, 0, stream>>>(nte, pos, means, stds, mult, biat,
                                                    b1, b2, w1b, w2b, maskArr, sumEdge, out);
    merge_kernel<<<(B_*N_)/16, 256, 0, stream>>>(sumEdge, wpb, bp, maskArr, out);
}
